// GIN_1408749273893
// MI455X (gfx1250) — compile-verified, matrix operand order checked by reading, not run on hardware
//
#include <hip/hip_runtime.h>
#include <hip/hip_bf16.h>

#define N_NODES 50000
#define N_EDGES 500000
#define D_IN 768
#define D_H 512
#define BN_EPS 1e-5f

typedef __attribute__((ext_vector_type(16))) __bf16 v16bf;
typedef __attribute__((ext_vector_type(8)))  __bf16 v8bf;
typedef __attribute__((ext_vector_type(8)))  float  v8f;

// ---------- helpers ----------
// Pack two f32 into two bf16 (RTNE) -- clang lowers fptrunc-to-bf16 to the
// hardware cvt path on gfx1250 (v_cvt_pk_bf16_f32 class) instead of the long
// manual integer-rounding sequence.
__device__ __forceinline__ unsigned int pk_bf16(float a, float b) {
    union { __bf16 h[2]; unsigned int u; } v;
    v.h[0] = (__bf16)a;
    v.h[1] = (__bf16)b;
    return v.u;
}

__device__ __forceinline__ unsigned short f32_to_bf16_s(float f) {
    union { __bf16 h; unsigned short u; } v;
    v.h = (__bf16)f;
    return v.u;
}

__device__ __forceinline__ v16bf load_frag(const unsigned short* rowBase, int hoff) {
    // rowBase points at the start of a 40-halfword LDS row (80B, 16B aligned).
    // hoff = 0 (lanes 0-15) or 8 (lanes 16-31): K 0-7/16-23 vs K 8-15/24-31.
    union { v16bf v; v8bf h[2]; } u;
    u.h[0] = *(const v8bf*)(rowBase + hoff);        // ds_load_b128
    u.h[1] = *(const v8bf*)(rowBase + hoff + 16);   // ds_load_b128
    return u.v;
}

// ---------- prep kernels ----------
// Wt[n*K + k] = bf16(W[k*N + n])  (transpose so B fragments are contiguous in K)
__global__ void gin_pack_weights(const float* __restrict__ W, unsigned short* __restrict__ Wt,
                                 int K, int N) {
    int id = blockIdx.x * blockDim.x + threadIdx.x;
    if (id >= K * N) return;
    int n = id / K, k = id % K;
    Wt[id] = f32_to_bf16_s(W[(size_t)k * N + n]);
}

// Fold bias (+ optional BN eval stats) into per-column scale/shift
__global__ void gin_fold_epilogue(const float* __restrict__ b,
                                  const float* __restrict__ gamma, const float* __restrict__ beta,
                                  const float* __restrict__ mean,  const float* __restrict__ var,
                                  float* __restrict__ s, float* __restrict__ t, int useBN) {
    int j = blockIdx.x * blockDim.x + threadIdx.x;
    if (j >= D_H) return;
    if (useBN) {
        float inv = rsqrtf(var[j] + BN_EPS);
        float g = gamma[j] * inv;
        s[j] = g;
        t[j] = beta[j] + (b[j] - mean[j]) * g;
    } else {
        s[j] = 1.0f;
        t[j] = b[j];
    }
}

__global__ void gin_copy_f4(const float4* __restrict__ in, float4* __restrict__ out, long n4) {
    long i = (long)blockIdx.x * blockDim.x + threadIdx.x;
    long stride = (long)gridDim.x * blockDim.x;
    for (; i < n4; i += stride) out[i] = in[i];
}

// ---------- edge scatter-add: agg[dst] += x[src] ----------
// One block per edge; src/dst become scalar loads (uniform across block).
template<int D>
__global__ void gin_scatter(const float* __restrict__ xin,
                            const int* __restrict__ src, const int* __restrict__ dst,
                            float* __restrict__ agg) {
    int e = blockIdx.x;
    int s = src[e];
    int d = dst[e];
    int t = threadIdx.x;                      // blockDim.x == D/4
    float4 v = ((const float4*)(xin + (size_t)s * D))[t];
    float* o = agg + (size_t)d * D + t * 4;
    __hip_atomic_fetch_add(o + 0, v.x, __ATOMIC_RELAXED, __HIP_MEMORY_SCOPE_AGENT);
    __hip_atomic_fetch_add(o + 1, v.y, __ATOMIC_RELAXED, __HIP_MEMORY_SCOPE_AGENT);
    __hip_atomic_fetch_add(o + 2, v.z, __ATOMIC_RELAXED, __HIP_MEMORY_SCOPE_AGENT);
    __hip_atomic_fetch_add(o + 3, v.w, __ATOMIC_RELAXED, __HIP_MEMORY_SCOPE_AGENT);
}

// ---------- WMMA GEMM: C[M,512] = relu(A[M,K] @ W[K,512] * s + t) ----------
// Block = 256 threads = 8 waves. Tile 128(M) x 64(N); each wave does 32x32
// via 4 x v_wmma_f32_16x16x32_bf16 per K-step of 32. Double-buffered LDS:
// one barrier per K-step, next tile's global loads overlap current WMMAs.
#define ASTR 40   // LDS row stride in halfwords (80B, conflict-free, 16B aligned)

template<int K>
__global__ __launch_bounds__(256) void gin_gemm_bf16(
        const float* __restrict__ A, const unsigned short* __restrict__ Wt,
        const float* __restrict__ sArr, const float* __restrict__ tArr,
        float* __restrict__ C, int M) {
    __shared__ unsigned short As[2][128 * ASTR];
    __shared__ unsigned short Bs[2][64 * ASTR];

    const int tid = threadIdx.x;
    const int rowBase = blockIdx.x * 128;
    const int colBase = blockIdx.y * 64;

    const int wid  = tid >> 5;
    const int lane = tid & 31;
    const int wm = (wid & 3) * 32;     // wave M offset in tile
    const int wn = (wid >> 2) * 32;    // wave N offset in tile
    const int lr = lane & 15;
    const int hoff = (lane & 16) ? 8 : 0;

    // stage tile kt into LDS buffer b
    auto stage = [&](int kt, int b) {
        // A: 128 rows x 32 f32 -> bf16
        #pragma unroll
        for (int i = 0; i < 4; ++i) {
            int idx = tid + i * 256;          // 0..1023
            int r  = idx >> 3;                // 0..127
            int c4 = idx & 7;                 // float4 index within 32-wide row
            int grow = rowBase + r;
            float4 v = make_float4(0.f, 0.f, 0.f, 0.f);
            if (grow < M)
                v = *(const float4*)(A + (size_t)grow * K + kt * 32 + c4 * 4);
            uint2 p;
            p.x = pk_bf16(v.x, v.y);
            p.y = pk_bf16(v.z, v.w);
            *(uint2*)(&As[b][r * ASTR + c4 * 4]) = p;
        }
        // B: 64 cols x 32 bf16 (pre-transposed, contiguous in K)
        int r = tid >> 2;                     // 0..63 : output column
        int c = tid & 3;                      // 16B chunk
        uint4 v = *(const uint4*)(Wt + (size_t)(colBase + r) * K + kt * 32 + c * 8);
        *(uint4*)(&Bs[b][r * ASTR + c * 8]) = v;
    };

    v8f acc00 = {}, acc01 = {}, acc10 = {}, acc11 = {};
    constexpr int KT = K / 32;

    stage(0, 0);
    __syncthreads();

    int b = 0;
    for (int kt = 0; kt < KT; ++kt) {
        if (kt + 1 < KT) stage(kt + 1, b ^ 1);   // fill other buffer under compute

        v16bf a0 = load_frag(&As[b][(wm      + lr) * ASTR], hoff);
        v16bf a1 = load_frag(&As[b][(wm + 16 + lr) * ASTR], hoff);
        v16bf b0 = load_frag(&Bs[b][(wn      + lr) * ASTR], hoff);
        v16bf b1 = load_frag(&Bs[b][(wn + 16 + lr) * ASTR], hoff);

        acc00 = __builtin_amdgcn_wmma_f32_16x16x32_bf16(false, a0, false, b0, (short)0, acc00, false, false);
        acc01 = __builtin_amdgcn_wmma_f32_16x16x32_bf16(false, a0, false, b1, (short)0, acc01, false, false);
        acc10 = __builtin_amdgcn_wmma_f32_16x16x32_bf16(false, a1, false, b0, (short)0, acc10, false, false);
        acc11 = __builtin_amdgcn_wmma_f32_16x16x32_bf16(false, a1, false, b1, (short)0, acc11, false, false);

        __syncthreads();                         // b^1 fully staged; b fully consumed
        b ^= 1;
    }

    // epilogue: C layout VGPR r -> M = r (lanes 0-15) / r+8 (lanes 16-31), N = lane&15
    const int mAdd = (lane & 16) ? 8 : 0;
    const v8f* accs[2][2] = { {&acc00, &acc01}, {&acc10, &acc11} };
    #pragma unroll
    for (int mi = 0; mi < 2; ++mi) {
        #pragma unroll
        for (int ni = 0; ni < 2; ++ni) {
            int col = colBase + wn + ni * 16 + lr;
            float sc = sArr[col];
            float tb = tArr[col];
            const v8f a = *accs[mi][ni];
            #pragma unroll
            for (int r = 0; r < 8; ++r) {
                int row = rowBase + wm + mi * 16 + r + mAdd;
                if (row < M) {
                    float v = a[r] * sc + tb;
                    C[(size_t)row * D_H + col] = v > 0.f ? v : 0.f;
                }
            }
        }
    }
}

// ---------- output head: out[n,0:2] = h2[n,:] @ Wout + bout (wave per node) ----------
__global__ void gin_out_head(const float* __restrict__ h, const float* __restrict__ Wout,
                             const float* __restrict__ bout, float* __restrict__ out) {
    int gid  = blockIdx.x * blockDim.x + threadIdx.x;
    int node = gid >> 5;
    int lane = gid & 31;
    if (node >= N_NODES) return;
    float a0 = 0.f, a1 = 0.f;
    const float* row = h + (size_t)node * D_H;
    for (int k = lane; k < D_H; k += 32) {
        float v = row[k];
        a0 += v * Wout[2 * k + 0];
        a1 += v * Wout[2 * k + 1];
    }
    #pragma unroll
    for (int off = 16; off > 0; off >>= 1) {
        a0 += __shfl_down(a0, off, 32);
        a1 += __shfl_down(a1, off, 32);
    }
    if (lane == 0) {
        out[2 * node + 0] = a0 + bout[0];
        out[2 * node + 1] = a1 + bout[1];
    }
}

// ---------- launch ----------
extern "C" void kernel_launch(void* const* d_in, const int* in_sizes, int n_in,
                              void* d_out, int out_size, void* d_ws, size_t ws_size,
                              hipStream_t stream) {
    const float* x    = (const float*)d_in[0];
    const int*   ei   = (const int*)d_in[1];
    const int*   srcI = ei;
    const int*   dstI = ei + N_EDGES;
    const float* W1a = (const float*)d_in[2];  const float* b1a = (const float*)d_in[3];
    const float* g1  = (const float*)d_in[4];  const float* be1 = (const float*)d_in[5];
    const float* m1  = (const float*)d_in[6];  const float* v1  = (const float*)d_in[7];
    const float* W1b = (const float*)d_in[8];  const float* b1b = (const float*)d_in[9];
    const float* W2a = (const float*)d_in[10]; const float* b2a = (const float*)d_in[11];
    const float* g2  = (const float*)d_in[12]; const float* be2 = (const float*)d_in[13];
    const float* m2  = (const float*)d_in[14]; const float* v2  = (const float*)d_in[15];
    const float* W2b = (const float*)d_in[16]; const float* b2b = (const float*)d_in[17];
    const float* Wout = (const float*)d_in[18]; const float* bout = (const float*)d_in[19];
    float* out = (float*)d_out;

    // workspace carve (all offsets 256B aligned)
    char* w = (char*)d_ws;
    size_t off = 0;
    auto carve = [&](size_t bytes) { void* p = w + off; off += (bytes + 255) & ~(size_t)255; return p; };
    unsigned short* Wt1a = (unsigned short*)carve((size_t)D_IN * D_H * 2);
    unsigned short* Wt1b = (unsigned short*)carve((size_t)D_H * D_H * 2);
    unsigned short* Wt2a = (unsigned short*)carve((size_t)D_H * D_H * 2);
    unsigned short* Wt2b = (unsigned short*)carve((size_t)D_H * D_H * 2);
    float* sAll = (float*)carve(4 * D_H * sizeof(float));
    float* tAll = (float*)carve(4 * D_H * sizeof(float));
    float* agg  = (float*)carve((size_t)N_NODES * D_IN * sizeof(float)); // also agg2 (D_H)
    float* bufA = (float*)carve((size_t)N_NODES * D_H * sizeof(float));  // t1 / t2
    float* bufB = (float*)carve((size_t)N_NODES * D_H * sizeof(float));  // h1 / h2
    (void)ws_size; (void)n_in; (void)in_sizes; (void)out_size;

    // 0) weight pack + epilogue fold
    gin_pack_weights<<<(D_IN * D_H + 255) / 256, 256, 0, stream>>>(W1a, Wt1a, D_IN, D_H);
    gin_pack_weights<<<(D_H * D_H + 255) / 256, 256, 0, stream>>>(W1b, Wt1b, D_H, D_H);
    gin_pack_weights<<<(D_H * D_H + 255) / 256, 256, 0, stream>>>(W2a, Wt2a, D_H, D_H);
    gin_pack_weights<<<(D_H * D_H + 255) / 256, 256, 0, stream>>>(W2b, Wt2b, D_H, D_H);
    gin_fold_epilogue<<<2, 256, 0, stream>>>(b1a, g1, be1, m1, v1, sAll + 0 * D_H, tAll + 0 * D_H, 1);
    gin_fold_epilogue<<<2, 256, 0, stream>>>(b1b, b1b, b1b, b1b, b1b, sAll + 1 * D_H, tAll + 1 * D_H, 0);
    gin_fold_epilogue<<<2, 256, 0, stream>>>(b2a, g2, be2, m2, v2, sAll + 2 * D_H, tAll + 2 * D_H, 1);
    gin_fold_epilogue<<<2, 256, 0, stream>>>(b2b, b2b, b2b, b2b, b2b, sAll + 3 * D_H, tAll + 3 * D_H, 0);

    dim3 gGemm((N_NODES + 127) / 128, D_H / 64);

    // 1) layer 1: agg = x + segment_sum(x[src] -> dst)
    gin_copy_f4<<<4096, 256, 0, stream>>>((const float4*)x, (float4*)agg,
                                          (long)N_NODES * D_IN / 4);
    gin_scatter<D_IN><<<N_EDGES, D_IN / 4, 0, stream>>>(x, srcI, dstI, agg);
    gin_gemm_bf16<D_IN><<<gGemm, 256, 0, stream>>>(agg, Wt1a, sAll + 0 * D_H, tAll + 0 * D_H, bufA, N_NODES);
    gin_gemm_bf16<D_H><<<gGemm, 256, 0, stream>>>(bufA, Wt1b, sAll + 1 * D_H, tAll + 1 * D_H, bufB, N_NODES);

    // 2) layer 2: agg = h1 + segment_sum(h1[src] -> dst)  (reuses agg buffer, D_H wide)
    gin_copy_f4<<<4096, 256, 0, stream>>>((const float4*)bufB, (float4*)agg,
                                          (long)N_NODES * D_H / 4);
    gin_scatter<D_H><<<N_EDGES, D_H / 4, 0, stream>>>(bufB, srcI, dstI, agg);
    gin_gemm_bf16<D_H><<<gGemm, 256, 0, stream>>>(agg, Wt2a, sAll + 2 * D_H, tAll + 2 * D_H, bufA, N_NODES);
    gin_gemm_bf16<D_H><<<gGemm, 256, 0, stream>>>(bufA, Wt2b, sAll + 3 * D_H, tAll + 3 * D_H, bufB, N_NODES);

    // 3) output head
    gin_out_head<<<(N_NODES * 32 + 255) / 256, 256, 0, stream>>>(bufB, Wout, bout, out);
}
